// CrossAttn_21543555957465
// MI455X (gfx1250) — matrix operand order repaired and checked by reading, not compiled, not run on hardware
//
#include <hip/hip_runtime.h>
#include <hip/hip_bf16.h>
#include <math.h>

// CDNA5 / gfx1250: wave32, WMMA 16x16x32 bf16 with f32 accumulate.
typedef __attribute__((ext_vector_type(16))) __bf16 v16bf;
typedef __attribute__((ext_vector_type(8)))  float  v8f;

#define HDIM 32   // head dim == WMMA K for bf16 -> one WMMA per 16x16 score tile
#define KBLK 128  // keys per flash block: amortizes softmax butterfly 4x

static __device__ __forceinline__ v8f wmma_bf16(v16bf a, v16bf b, v8f c) {
  // (neg_a, A, neg_b, B, c_mod, C, reuse_a, reuse_b)
  return __builtin_amdgcn_wmma_f32_16x16x32_bf16(false, a, false, b, (short)0, c,
                                                 false, false);
}

union BV { v16bf v; uint4 u[2]; };

// ---------------------------------------------------------------------------
// 1x1-conv projection as WMMA GEMM: out[o,t] = sum_c W[o,c]*x[b,c,t] + bias[o]
// One wave computes a 16(o) x 64(t) tile; K loop over C in steps of 32.
// MODE 0: bf16 out, [B,C,T]            (K projection: natural B-layout rows)
// MODE 1: bf16 out, [B,H,T,D], *scale  (Q with 1/sqrt(D) folded, V with 1.0)
// MODE 2: f32  out, [B,C,T], * mask    (final projection straight to d_out)
// ---------------------------------------------------------------------------
template<int MODE, typename Tin>
__global__ void
__launch_bounds__(32)
__attribute__((amdgpu_waves_per_eu(1, 8)))
proj1x1_kernel(const Tin* __restrict__ x,
               const float* __restrict__ W,
               const float* __restrict__ bias,
               const unsigned char* __restrict__ mask,
               void* __restrict__ outp,
               float scaleOut, int B, int C, int T) {
  const int lane = threadIdx.x;
  const int b  = blockIdx.z;
  const int o0 = blockIdx.y * 16;
  const int t0 = blockIdx.x * 64;
  const int rowm = lane & 15;
  const int kb0 = (lane < 16) ? 0 : 8;   // A-layout K bases (16-bit A 16x32)
  const int kb1 = kb0 + 16;

  v8f acc[4] = {};

  for (int c0 = 0; c0 < C; c0 += 32) {
    // A tile: W rows (f32 -> bf16), lane holds row o0+rowm, split K pattern
    const float* wr = W + (size_t)(o0 + rowm) * C + c0;
    v16bf a;
#pragma unroll
    for (int i = 0; i < 8; ++i) {
      a[i]     = (__bf16)wr[kb0 + i];
      a[8 + i] = (__bf16)wr[kb1 + i];
    }
    // B tiles: lane = t column (t0+j*16+lane%16), split K pattern over c
    const Tin* xr = x + ((size_t)b * C + c0) * T + t0 + (lane & 15);
#pragma unroll
    for (int j = 0; j < 4; ++j) {
      v16bf bm;
#pragma unroll
      for (int i = 0; i < 8; ++i) {
        bm[i]     = (__bf16)(float)xr[(size_t)(kb0 + i) * T + j * 16];
        bm[8 + i] = (__bf16)(float)xr[(size_t)(kb1 + i) * T + j * 16];
      }
      acc[j] = wmma_bf16(a, bm, acc[j]);
    }
  }

  const int ncol = lane & 15;
  const int mhi  = (lane >> 4) * 8;   // C/D layout: lanes16-31 hold rows 8..15
#pragma unroll
  for (int r = 0; r < 8; ++r) {
    const int o = o0 + r + mhi;
    const float bvv = bias[o];
#pragma unroll
    for (int j = 0; j < 4; ++j) {
      const int t = t0 + j * 16 + ncol;
      const float val = acc[j][r] + bvv;
      if (MODE == 0) {
        ((__bf16*)outp)[((size_t)b * C + o) * T + t] = (__bf16)(val * scaleOut);
      } else if (MODE == 1) {
        const int h = o >> 5, d = o & 31;
        ((__bf16*)outp)[(((size_t)b * (C >> 5) + h) * T + t) * HDIM + d] =
            (__bf16)(val * scaleOut);
      } else {
        const float mv = mask[(size_t)b * T + t] ? 1.f : 0.f;
        ((float*)outp)[((size_t)b * C + o) * T + t] = val * mv;
      }
    }
  }
}

// ---------------------------------------------------------------------------
// Fused flash attention: one wave per (b, h, 16-query tile).
// 128-key blocks: 8 S-WMMAs, ONE max butterfly + ONE sum butterfly per row,
// then 8 PV-WMMAs. P goes C/D-layout -> LDS (4 KB) -> A-layout.
// __launch_bounds__(32): single-wave workgroup; lifts VGPR cap so the 8 live
// S tiles (64 VGPRs) + acc + Q stay in registers (no scratch spills).
// ---------------------------------------------------------------------------
__global__ void
__launch_bounds__(32)
__attribute__((amdgpu_waves_per_eu(1, 8)))
flash_attn_kernel(const __bf16* __restrict__ Qh,  // [B,H,T,D], scale folded
                  const __bf16* __restrict__ Kt,  // [B,C,T]
                  const __bf16* __restrict__ Vh,  // [B,H,T,D]
                  const unsigned char* __restrict__ mask, // [B,T]
                  __bf16* __restrict__ Oa,        // [B,C,T]
                  int B, int H, int T) {
  __shared__ __bf16 pLds[16 * KBLK];   // 4 KB per wave
  const int lane = threadIdx.x;
  const int b = blockIdx.z, h = blockIdx.y;
  const int q0 = blockIdx.x * 16;
  const int rowm = lane & 15;
  const int kb0 = (lane < 16) ? 0 : 8;
  const int kb1 = kb0 + 16;
  const int ncol = lane & 15;
  const int mhi  = (lane >> 4) * 8;

  // Q tile, A-layout: lane holds query row q0+rowm, K pattern over D
  BV aq;
  const __bf16* qr = Qh + (((size_t)b * H + h) * T + (q0 + rowm)) * HDIM;
  aq.u[0] = *(const uint4*)(qr + kb0);
  aq.u[1] = *(const uint4*)(qr + kb1);

  float M[8], L[8];
#pragma unroll
  for (int r = 0; r < 8; ++r) { M[r] = -3.0e38f; L[r] = 0.f; }
  v8f acc0 = {}, acc1 = {};

  const __bf16* kbase = Kt + (((size_t)b * H + h) * HDIM + lane) * T; // row c = h*32+lane
  const __bf16* vbase = Vh + ((size_t)b * H + h) * T * HDIM;
  const unsigned char* mrow = mask + (size_t)b * T;

  for (int kc = 0; kc < T; kc += KBLK) {
    // ---- Scores: 8 tiles of 16 keys each (K = D = 32 per WMMA) ----
    v8f S[8];
    const v8f z = {};
#pragma unroll
    for (int tp = 0; tp < 4; ++tp) {
      BV bk0, bk1;                       // lane = key column, split K pattern over d
      const __bf16* kr = Kt + ((size_t)b * H + h) * HDIM * T + kc + tp * 32 + ncol;
#pragma unroll
      for (int i = 0; i < 8; ++i) {
        bk0.v[i]     = kr[(size_t)(kb0 + i) * T];
        bk0.v[8 + i] = kr[(size_t)(kb1 + i) * T];
        bk1.v[i]     = kr[(size_t)(kb0 + i) * T + 16];
        bk1.v[8 + i] = kr[(size_t)(kb1 + i) * T + 16];
      }
      S[2 * tp]     = wmma_bf16(aq.v, bk0.v, z);
      S[2 * tp + 1] = wmma_bf16(aq.v, bk1.v, z);
    }

    bool mk[8];
#pragma unroll
    for (int tt = 0; tt < 8; ++tt) mk[tt] = mrow[kc + 16 * tt + ncol] != 0;

    // ---- Online softmax over the whole 128-key block ----
#pragma unroll
    for (int r = 0; r < 8; ++r) {
      float s[8];
      float cm = -3.0e38f;
#pragma unroll
      for (int tt = 0; tt < 8; ++tt) {
        s[tt] = mk[tt] ? S[tt][r] : -3.0e38f;
        cm = fmaxf(cm, s[tt]);
      }
#pragma unroll
      for (int off = 1; off < 16; off <<= 1)
        cm = fmaxf(cm, __shfl_xor(cm, off, 32));
      const float newM  = fmaxf(M[r], cm);
      const float alpha = __expf(M[r] - newM);
      float ps = 0.f;
      const int mr = r + mhi;            // C/D row index
#pragma unroll
      for (int tt = 0; tt < 8; ++tt) {
        const float p = mk[tt] ? __expf(s[tt] - newM) : 0.f;
        ps += p;
        pLds[mr * KBLK + 16 * tt + ncol] = (__bf16)p;
      }
#pragma unroll
      for (int off = 1; off < 16; off <<= 1)
        ps += __shfl_xor(ps, off, 32);
      L[r] = L[r] * alpha + ps;
      M[r] = newM;
      acc0[r] *= alpha;
      acc1[r] *= alpha;
    }
    __syncthreads();

    // ---- P.V over four 32-key sub-chunks ----
#pragma unroll
    for (int cc = 0; cc < 4; ++cc) {
      BV ap;                             // P in A-layout from LDS row-major
      ap.u[0] = *(const uint4*)(&pLds[rowm * KBLK + cc * 32 + kb0]);
      ap.u[1] = *(const uint4*)(&pLds[rowm * KBLK + cc * 32 + kb1]);
      BV bv0, bv1;                       // lane = d column, split K pattern over keys
      const __bf16* vr = vbase + (size_t)(kc + cc * 32) * HDIM + ncol;
#pragma unroll
      for (int i = 0; i < 8; ++i) {
        bv0.v[i]     = vr[(kb0 + i) * HDIM];
        bv0.v[8 + i] = vr[(kb1 + i) * HDIM];
        bv1.v[i]     = vr[(kb0 + i) * HDIM + 16];
        bv1.v[8 + i] = vr[(kb1 + i) * HDIM + 16];
      }
      acc0 = wmma_bf16(ap.v, bv0.v, acc0);   // d = 0..15
      acc1 = wmma_bf16(ap.v, bv1.v, acc1);   // d = 16..31
    }
    __syncthreads();

    if (kc + KBLK < T) {                 // global_prefetch_b8 next K/V block
      __builtin_prefetch(kbase + kc + KBLK, 0, 1);
      __builtin_prefetch(vbase + (size_t)(kc + KBLK + lane) * HDIM, 0, 1);
    }
  }

  // Normalize and write attention output as [B,C,T] bf16 (B-layout rows for
  // the final projection GEMM).
#pragma unroll
  for (int r = 0; r < 8; ++r) {
    const float inv = 1.0f / L[r];
    const int t = q0 + r + mhi;
    Oa[(((size_t)b * H + h) * HDIM + ncol) * T + t]      = (__bf16)(acc0[r] * inv);
    Oa[(((size_t)b * H + h) * HDIM + 16 + ncol) * T + t] = (__bf16)(acc1[r] * inv);
  }
}

__global__ void __launch_bounds__(256)
mask_to_float_kernel(const unsigned char* __restrict__ m,
                     float* __restrict__ o, int n) {
  int i = blockIdx.x * 256 + threadIdx.x;
  if (i < n) o[i] = m[i] ? 1.f : 0.f;
}

extern "C" void kernel_launch(void* const* d_in, const int* in_sizes, int n_in,
                              void* d_out, int out_size, void* d_ws, size_t ws_size,
                              hipStream_t stream) {
  (void)in_sizes; (void)n_in; (void)ws_size;
  const int B = 2, C = 512, T = 2048, H = 16;
  const float scale = 0.17677669529663687f;  // 1/sqrt(32), folded into Q proj

  const float* q  = (const float*)d_in[0];
  const float* k  = (const float*)d_in[1];
  const float* v  = (const float*)d_in[2];
  const unsigned char* mask = (const unsigned char*)d_in[3];
  const float* Wq = (const float*)d_in[4];
  const float* bq = (const float*)d_in[5];
  const float* Wk = (const float*)d_in[6];
  const float* bk = (const float*)d_in[7];
  const float* Wv = (const float*)d_in[8];
  const float* bv = (const float*)d_in[9];
  const float* Wp = (const float*)d_in[10];
  const float* bp = (const float*)d_in[11];

  char* ws = (char*)d_ws;
  const size_t planeBytes = (size_t)B * C * T * sizeof(__bf16);  // 4 MB
  __bf16* Qh = (__bf16*)(ws);                      // [B,H,T,D]
  __bf16* Kt = (__bf16*)(ws + planeBytes);         // [B,C,T]
  __bf16* Vh = (__bf16*)(ws + 2 * planeBytes);     // [B,H,T,D]
  __bf16* Oa = (__bf16*)(ws + 3 * planeBytes);     // [B,C,T]

  dim3 gp(T / 64, C / 16, B);
  dim3 wv(32);
  proj1x1_kernel<1, float><<<gp, wv, 0, stream>>>(q, Wq, bq, mask, Qh, scale, B, C, T);
  proj1x1_kernel<0, float><<<gp, wv, 0, stream>>>(k, Wk, bk, mask, Kt, 1.0f, B, C, T);
  proj1x1_kernel<1, float><<<gp, wv, 0, stream>>>(v, Wv, bv, mask, Vh, 1.0f, B, C, T);

  dim3 ga(T / 16, H, B);
  flash_attn_kernel<<<ga, wv, 0, stream>>>(Qh, Kt, Vh, mask, Oa, B, H, T);

  proj1x1_kernel<2, __bf16><<<gp, wv, 0, stream>>>(Oa, Wp, bp, mask, (float*)d_out,
                                                   1.0f, B, C, T);

  // Second tuple output: mask, appended flat after out (B*C*T floats).
  const int BCT = B * C * T, BT = B * T;
  if (out_size >= BCT + BT) {
    mask_to_float_kernel<<<(BT + 255) / 256, 256, 0, stream>>>(
        mask, (float*)d_out + BCT, BT);
  }
}
